// TCNAttentionBlock_5317169513057
// MI455X (gfx1250) — matrix operand order repaired and checked by reading, not compiled, not run on hardware
//
#include <hip/hip_runtime.h>
#include <hip/hip_bf16.h>
#include <math.h>

#define NB 8
#define TT 1024
#define EE 512
#define HH 8
#define DD 64
#define PP 16
#define KT (TT + PP) /* 1040 keys incl. persistent */

typedef __attribute__((ext_vector_type(16))) _Float16 v16h;
typedef __attribute__((ext_vector_type(8)))  float    v8f;

union AF16 { v16h v; _Float16 h[16]; };
union CF32 { v8f  v; float     f[8]; };

__device__ __forceinline__ v8f zerov8() {
  v8f z;
#pragma unroll
  for (int i = 0; i < 8; ++i) z[i] = 0.f;
  return z;
}

__device__ __forceinline__ v8f wmma_f16(v16h a, v16h b, v8f c) {
  // D = A(16x32 f16) * B(32x16 f16) + C(16x16 f32)
  return __builtin_amdgcn_wmma_f32_16x16x32_f16(false, a, false, b, (short)0, c,
                                                false, false);
}

// A operand: 16(M) x 32(K) tile, row-major with row stride `ld` (halfs).
// Lane layout: contiguous k = 16*(lane/16) + s (matches B loaders).
__device__ __forceinline__ v16h loadA(const _Float16* b, int ld, int lane) {
  int m = lane & 15, hi = lane >> 4;
  AF16 a;
#pragma unroll
  for (int i = 0; i < 8; ++i) {
    int k = 2 * i + hi * 16;
    a.h[2 * i]     = b[m * ld + k];
    a.h[2 * i + 1] = b[m * ld + k + 1];
  }
  return a.v;
}

// B operand from an N-major source: element(k,n) = b[n*ldn + k] (i.e. source
// holds the transposed B, e.g. weight rows). 32(K) x 16(N).
__device__ __forceinline__ v16h loadBT(const _Float16* b, int ldn, int lane) {
  int n = lane & 15, kb = (lane >> 4) * 16;
  AF16 f;
#pragma unroll
  for (int i = 0; i < 8; ++i) {
    f.h[2 * i]     = b[n * ldn + kb + 2 * i];
    f.h[2 * i + 1] = b[n * ldn + kb + 2 * i + 1];
  }
  return f.v;
}

// B operand from a K-major source with guard: element(k,n) = b[k*ldk + n],
// rows with k >= kmax read as zero (tail of the 1040-key dimension).
__device__ __forceinline__ v16h loadBK(const _Float16* b, int ldk, int lane,
                                       int kmax) {
  int n = lane & 15, kb = (lane >> 4) * 16;
  AF16 f;
#pragma unroll
  for (int i = 0; i < 16; ++i) {
    int k = kb + i;
    f.h[i] = (k < kmax) ? b[(size_t)k * ldk + n] : (_Float16)0.f;
  }
  return f.v;
}

// ---------------------------------------------------------------- K1: QKV ---
__global__ void __launch_bounds__(256) k_qkv(const float* __restrict__ x,
                                             const float* __restrict__ Wq,
                                             const float* __restrict__ Wk,
                                             const float* __restrict__ Wv,
                                             _Float16* __restrict__ qo,
                                             _Float16* __restrict__ ko,
                                             _Float16* __restrict__ vo) {
  __shared__ _Float16 wq[64 * 64], wk[64 * 64], wv[64 * 64];
  __shared__ _Float16 al[128 * 64];
  int tid = threadIdx.x, lane = tid & 31, wid = tid >> 5;
  int m0 = blockIdx.x * 128; // row = (n*T+t)*H + h
  for (int i = tid; i < 4096; i += 256) {
    wq[i] = (_Float16)Wq[i];
    wk[i] = (_Float16)Wk[i];
    wv[i] = (_Float16)Wv[i];
  }
  for (int i = tid; i < 128 * 64; i += 256)
    al[i] = (_Float16)x[(size_t)(m0 + (i >> 6)) * 64 + (i & 63)];
  __syncthreads();

  const _Float16* ab = al + wid * 16 * 64;
  v16h a0 = loadA(ab, 64, lane);
  v16h a1 = loadA(ab + 32, 64, lane);
  int hi = lane >> 4, nn = lane & 15;

#pragma unroll
  for (int w = 0; w < 3; ++w) {
    const _Float16* wl = (w == 0) ? wq : (w == 1) ? wk : wv;
#pragma unroll
    for (int nt = 0; nt < 4; ++nt) {
      CF32 acc;
      acc.v = zerov8();
      acc.v = wmma_f16(a0, loadBT(wl + nt * 16 * 64, 64, lane), acc.v);
      acc.v = wmma_f16(a1, loadBT(wl + nt * 16 * 64 + 32, 64, lane), acc.v);
#pragma unroll
      for (int r = 0; r < 8; ++r) {
        int gr = m0 + wid * 16 + r + 8 * hi;
        int dout = nt * 16 + nn;
        if (w == 0) {
          qo[(size_t)gr * 64 + dout] = (_Float16)acc.f[r];
        } else {
          int hh = gr & 7, nt_ = gr >> 3;
          int t = nt_ & (TT - 1), nb = nt_ >> 10;
          size_t ki = (((size_t)(nb * KT + t)) * HH + hh) * DD + dout;
          if (w == 1) ko[ki] = (_Float16)acc.f[r];
          else        vo[ki] = (_Float16)acc.f[r];
        }
      }
    }
  }
}

// --------------------------------------------- K1b: persistent k/v append ---
__global__ void k_persist(const float* __restrict__ pk,
                          const float* __restrict__ pv,
                          _Float16* __restrict__ ko,
                          _Float16* __restrict__ vo) {
  int i = blockIdx.x * blockDim.x + threadIdx.x;
  if (i >= NB * PP * HH * DD) return;
  int d = i & 63, h = (i >> 6) & 7, p = (i >> 9) & 15, nb = i >> 13;
  size_t src = ((size_t)p * HH + h) * DD + d;
  size_t dst = (((size_t)nb * KT + TT + p) * HH + h) * DD + d;
  ko[dst] = (_Float16)pk[src];
  vo[dst] = (_Float16)pv[src];
}

// ---------------------------------------------------------- K2: attention ---
__global__ void __launch_bounds__(256) k_attn(const _Float16* __restrict__ q,
                                              const _Float16* __restrict__ k,
                                              const _Float16* __restrict__ v,
                                              const int* __restrict__ seqm,
                                              const float* __restrict__ pre_th,
                                              const float* __restrict__ post_th,
                                              _Float16* __restrict__ att) {
  __shared__ float    e_lds[8 * 256]; // per-head raw energy (+ alibi), 16x16
  __shared__ float    p_lds[8 * 256]; // per-head normalized probs, 16x16
  __shared__ _Float16 a_lds[8 * 512]; // per-wave post-mixed attn A tile 16x32
  int tid = threadIdx.x, lane = tid & 31, g = tid >> 5;
  int hi = lane >> 4, nn = lane & 15;
  int q0 = (blockIdx.x & 63) * 16;
  int nb = blockIdx.x >> 6;

  float pre[8], post[8];
#pragma unroll
  for (int h = 0; h < 8; ++h) {
    pre[h]  = pre_th[g * 8 + h];
    post[h] = post_th[g * 8 + h];
  }
  float slope = exp2f(-(float)(g + 1));
  const float rsqE = 0.04419417382415922f; // 1/sqrt(512)

  const _Float16* qb = q + (((size_t)(nb * TT + q0)) * HH + g) * DD;
  v16h aq0 = loadA(qb, 512, lane);      // d 0..31
  v16h aq1 = loadA(qb + 32, 512, lane); // d 32..63

  float Mr[8], Lr[8];
#pragma unroll
  for (int r = 0; r < 8; ++r) { Mr[r] = -3.0e38f; Lr[r] = 0.f; }

  // ---- pass 1: softmax statistics -------------------------------------
  for (int s = 0; s < 65; ++s) {
    int key0 = s * 16;
    const _Float16* kb = k + (((size_t)(nb * KT + key0)) * HH + g) * DD;
    CF32 e;
    e.v = zerov8();
    e.v = wmma_f16(aq0, loadBT(kb, 512, lane), e.v);
    e.v = wmma_f16(aq1, loadBT(kb + 32, 512, lane), e.v);
    int key = key0 + nn;
    int mv = (key < TT) ? seqm[nb * TT + key] : 1;
#pragma unroll
    for (int r = 0; r < 8; ++r) {
      int m = r + 8 * hi;
      float bias = (key < TT)
                       ? -fabsf((float)(q0 + m - key)) * slope
                       : 0.f;
      e_lds[g * 256 + m * 16 + nn] = e.f[r] + bias;
    }
    __syncthreads();
#pragma unroll
    for (int r = 0; r < 8; ++r) {
      int m = r + 8 * hi;
      float mix = 0.f;
#pragma unroll
      for (int h = 0; h < 8; ++h) mix += pre[h] * e_lds[h * 256 + m * 16 + nn];
      if (mv == 0) mix = -1e4f;
      float sc = mix * rsqE;
      float mnew = fmaxf(Mr[r], sc);
      Lr[r] = Lr[r] * __expf(Mr[r] - mnew) + __expf(sc - mnew);
      Mr[r] = mnew;
    }
    __syncthreads();
  }
  // combine stats across the 16 lanes sharing each row (half-wave groups)
#pragma unroll
  for (int off = 1; off < 16; off <<= 1) {
#pragma unroll
    for (int r = 0; r < 8; ++r) {
      float om = __shfl_xor(Mr[r], off, 32);
      float ol = __shfl_xor(Lr[r], off, 32);
      float mnew = fmaxf(Mr[r], om);
      Lr[r] = Lr[r] * __expf(Mr[r] - mnew) + ol * __expf(om - mnew);
      Mr[r] = mnew;
    }
  }
  float invL[8];
#pragma unroll
  for (int r = 0; r < 8; ++r) invL[r] = 1.f / fmaxf(Lr[r], 1e-20f);

  // ---- pass 2: normalize, post-mix, attn @ V --------------------------
  CF32 o[4];
#pragma unroll
  for (int nt = 0; nt < 4; ++nt) o[nt].v = zerov8();

  for (int s2 = 0; s2 < 33; ++s2) {
#pragma unroll
    for (int ss = 0; ss < 2; ++ss) {
      int step = s2 * 2 + ss;
      if (step < 65) {
        int key0 = step * 16;
        const _Float16* kb = k + (((size_t)(nb * KT + key0)) * HH + g) * DD;
        CF32 e;
        e.v = zerov8();
        e.v = wmma_f16(aq0, loadBT(kb, 512, lane), e.v);
        e.v = wmma_f16(aq1, loadBT(kb + 32, 512, lane), e.v);
        int key = key0 + nn;
        int mv = (key < TT) ? seqm[nb * TT + key] : 1;
#pragma unroll
        for (int r = 0; r < 8; ++r) {
          int m = r + 8 * hi;
          float bias = (key < TT)
                           ? -fabsf((float)(q0 + m - key)) * slope
                           : 0.f;
          e_lds[g * 256 + m * 16 + nn] = e.f[r] + bias;
        }
        __syncthreads();
#pragma unroll
        for (int r = 0; r < 8; ++r) {
          int m = r + 8 * hi;
          float mix = 0.f;
#pragma unroll
          for (int h = 0; h < 8; ++h)
            mix += pre[h] * e_lds[h * 256 + m * 16 + nn];
          if (mv == 0) mix = -1e4f;
          float sc = mix * rsqE;
          p_lds[g * 256 + m * 16 + nn] = __expf(sc - Mr[r]) * invL[r];
        }
        __syncthreads();
#pragma unroll
        for (int r = 0; r < 8; ++r) {
          int m = r + 8 * hi;
          float a2 = 0.f;
#pragma unroll
          for (int h = 0; h < 8; ++h)
            a2 += post[h] * p_lds[h * 256 + m * 16 + nn];
          a_lds[g * 512 + m * 32 + ss * 16 + nn] = (_Float16)a2;
        }
      } else { // uniform tail (s2==32, ss==1): zero fill, keep barriers
        __syncthreads();
#pragma unroll
        for (int r = 0; r < 8; ++r)
          p_lds[g * 256 + (r + 8 * hi) * 16 + nn] = 0.f;
        __syncthreads();
#pragma unroll
        for (int r = 0; r < 8; ++r)
          a_lds[g * 512 + (r + 8 * hi) * 32 + ss * 16 + nn] = (_Float16)0.f;
      }
    }
    // accumulate (16x32 probs) @ (32x64 V) for head g
    v16h a2 = loadA(a_lds + g * 512, 32, lane);
    int kmax = KT - s2 * 32; // >32 everywhere except the final tile (16)
    const _Float16* vb = v + (((size_t)(nb * KT + s2 * 32)) * HH + g) * DD;
#pragma unroll
    for (int nt = 0; nt < 4; ++nt)
      o[nt].v = wmma_f16(a2, loadBK(vb + nt * 16, 512, lane, kmax), o[nt].v);
  }

#pragma unroll
  for (int nt = 0; nt < 4; ++nt)
#pragma unroll
    for (int r = 0; r < 8; ++r) {
      int m = r + 8 * hi;
      att[((size_t)(nb * TT + q0 + m)) * EE + g * 64 + nt * 16 + nn] =
          (_Float16)o[nt].f[r];
    }
}

// ----------------------------------------- K3: Wo GEMM + bias + residual ---
__global__ void __launch_bounds__(256) k_wo(const _Float16* __restrict__ att,
                                            const float* __restrict__ Wo,
                                            const float* __restrict__ bo,
                                            const float* __restrict__ x,
                                            float* __restrict__ hout) {
  __shared__ _Float16 A[128 * 32];
  __shared__ _Float16 B[64 * 32];
  int tid = threadIdx.x, lane = tid & 31, w = tid >> 5;
  int m0 = (blockIdx.x >> 3) * 128, c0 = (blockIdx.x & 7) * 64;
  CF32 acc[4];
#pragma unroll
  for (int nt = 0; nt < 4; ++nt) acc[nt].v = zerov8();

  for (int ks = 0; ks < 16; ++ks) {
    int k0 = ks * 32;
    __syncthreads();
    for (int i = tid; i < 128 * 32; i += 256)
      A[i] = att[(size_t)(m0 + (i >> 5)) * 512 + k0 + (i & 31)];
    for (int i = tid; i < 64 * 32; i += 256)
      B[i] = (_Float16)Wo[(size_t)(c0 + (i >> 5)) * 512 + k0 + (i & 31)];
    __syncthreads();
    v16h a = loadA(A + w * 16 * 32, 32, lane);
#pragma unroll
    for (int nt = 0; nt < 4; ++nt)
      acc[nt].v = wmma_f16(a, loadBT(B + nt * 16 * 32, 32, lane), acc[nt].v);
  }
  int hi = lane >> 4, nn = lane & 15;
#pragma unroll
  for (int nt = 0; nt < 4; ++nt)
#pragma unroll
    for (int r = 0; r < 8; ++r) {
      int m = m0 + w * 16 + r + 8 * hi;
      int c = c0 + nt * 16 + nn;
      size_t idx = (size_t)m * 512 + c;
      hout[idx] = acc[nt].f[r] + bo[c] + x[idx];
    }
}

// --------------------------------- K4/K5: causal conv1d as implicit GEMM ---
__global__ void __launch_bounds__(256) k_conv(const float* __restrict__ in,
                                              const float* __restrict__ W,
                                              const float* __restrict__ bias,
                                              const int* __restrict__ seqm,
                                              float* __restrict__ out) {
  __shared__ _Float16 A[128 * 32];
  __shared__ _Float16 B[64 * 32];
  int tid = threadIdx.x, lane = tid & 31, w = tid >> 5;
  int m0 = (blockIdx.x >> 3) * 128, c0 = (blockIdx.x & 7) * 64;
  CF32 acc[4];
#pragma unroll
  for (int nt = 0; nt < 4; ++nt) acc[nt].v = zerov8();

  for (int j = 0; j < 3; ++j) {
    for (int ks = 0; ks < 16; ++ks) {
      int ci0 = ks * 32;
      __syncthreads();
      for (int i = tid; i < 128 * 32; i += 256) {
        int m = m0 + (i >> 5);
        int t = m & (TT - 1), nbk = m >> 10;
        int tp = t - 2 + j; // causal tap
        A[i] = (tp >= 0)
                   ? (_Float16)in[((size_t)((nbk << 10) | tp)) * 512 + ci0 +
                                  (i & 31)]
                   : (_Float16)0.f;
      }
      for (int i = tid; i < 64 * 32; i += 256)
        B[i] = (_Float16)W[((size_t)(c0 + (i >> 5)) * 512 + ci0 + (i & 31)) * 3 +
                           j];
      __syncthreads();
      v16h a = loadA(A + w * 16 * 32, 32, lane);
#pragma unroll
      for (int nt = 0; nt < 4; ++nt)
        acc[nt].v = wmma_f16(a, loadBT(B + nt * 16 * 32, 32, lane), acc[nt].v);
    }
  }
  int hi = lane >> 4, nn = lane & 15;
#pragma unroll
  for (int nt = 0; nt < 4; ++nt)
#pragma unroll
    for (int r = 0; r < 8; ++r) {
      int m = m0 + w * 16 + r + 8 * hi;
      int c = c0 + nt * 16 + nn;
      float val = fmaxf(acc[nt].f[r] + bias[c], 0.f);
      val *= (float)seqm[m]; // (N,T) flat == m
      out[(size_t)m * 512 + c] = val;
    }
}

// -------------------------------------- K6: residual ReLU + LayerNorm ------
__global__ void __launch_bounds__(256) k_ln(const float* __restrict__ y2,
                                            const float* __restrict__ h,
                                            const float* __restrict__ gg,
                                            const float* __restrict__ bb,
                                            float* __restrict__ out) {
  __shared__ float z[512];
  __shared__ float red[256];
  int tid = threadIdx.x;
  size_t base = (size_t)blockIdx.x * 512;
  float s = 0.f;
  for (int i = tid; i < 512; i += 256) {
    float val = fmaxf(y2[base + i] + h[base + i], 0.f);
    z[i] = val;
    s += val;
  }
  red[tid] = s;
  __syncthreads();
  for (int o = 128; o > 0; o >>= 1) {
    if (tid < o) red[tid] += red[tid + o];
    __syncthreads();
  }
  float mu = red[0] * (1.f / 512.f);
  __syncthreads();
  float s2 = 0.f;
  for (int i = tid; i < 512; i += 256) {
    float d = z[i] - mu;
    s2 += d * d;
  }
  red[tid] = s2;
  __syncthreads();
  for (int o = 128; o > 0; o >>= 1) {
    if (tid < o) red[tid] += red[tid + o];
    __syncthreads();
  }
  float rstd = rsqrtf(red[0] * (1.f / 512.f) + 1e-5f);
  for (int i = tid; i < 512; i += 256)
    out[base + i] = (z[i] - mu) * rstd * gg[i] + bb[i];
}

// ---------------------------------------------------------------------------
extern "C" void kernel_launch(void* const* d_in, const int* in_sizes, int n_in,
                              void* d_out, int out_size, void* d_ws,
                              size_t ws_size, hipStream_t stream) {
  const float* x       = (const float*)d_in[0];
  const int*   seqm    = (const int*)d_in[1];
  const float* Wq      = (const float*)d_in[2];
  const float* Wk      = (const float*)d_in[3];
  const float* Wv      = (const float*)d_in[4];
  const float* Wo      = (const float*)d_in[5];
  const float* bo      = (const float*)d_in[6];
  const float* pre_th  = (const float*)d_in[7];
  const float* post_th = (const float*)d_in[8];
  const float* pk      = (const float*)d_in[9];
  const float* pv      = (const float*)d_in[10];
  const float* c1W     = (const float*)d_in[11];
  const float* c1b     = (const float*)d_in[12];
  const float* c2W     = (const float*)d_in[13];
  const float* c2b     = (const float*)d_in[14];
  const float* lng     = (const float*)d_in[15];
  const float* lnb     = (const float*)d_in[16];

  char* ws = (char*)d_ws;
  // f16 q/k/v (k/v carry +P persistent keys)
  _Float16* qf   = (_Float16*)(ws + 0);            //  8,388,608 B
  _Float16* kf   = (_Float16*)(ws + 8388608);      //  8,519,680 B
  _Float16* vf   = (_Float16*)(ws + 16908288);     //  8,519,680 B
  _Float16* attf = (_Float16*)(ws + 25427968);     //  8,388,608 B
  float*    hbuf = (float*)(ws + 33816576);        // 16,777,216 B
  float*    y1   = (float*)(ws + 0);               // reuses dead q/k/v region
  float*    y2   = (float*)(ws + 50593792);        // 16,777,216 B (tot ~67MB)

  k_qkv<<<512, 256, 0, stream>>>(x, Wq, Wk, Wv, qf, kf, vf);
  k_persist<<<256, 256, 0, stream>>>(pk, pv, kf, vf);
  k_attn<<<512, 256, 0, stream>>>(qf, kf, vf, seqm, pre_th, post_th, attf);
  k_wo<<<512, 256, 0, stream>>>(attf, Wo, bo, x, hbuf);
  k_conv<<<512, 256, 0, stream>>>(hbuf, c1W, c1b, seqm, y1);
  k_conv<<<512, 256, 0, stream>>>(y1, c2W, c2b, seqm, y2);
  k_ln<<<8192, 256, 0, stream>>>(y2, hbuf, lng, lnb, (float*)d_out);
}
